// CustomLSTMCell_65609920413977
// MI455X (gfx1250) — matrix operand order repaired and checked by reading, not compiled, not run on hardware
//
#include <hip/hip_runtime.h>

// ---------------------------------------------------------------------------
// Types for CDNA5 WMMA
// ---------------------------------------------------------------------------
typedef __attribute__((ext_vector_type(16))) __bf16 v16bf;
typedef __attribute__((ext_vector_type(8)))  __bf16 v8bf;
typedef __attribute__((ext_vector_type(8)))  float  v8f;

#define B_SZ  8192
#define CIN_SZ 512
#define H_SZ  2048

// ---------------------------------------------------------------------------
// CDNA5 async global->LDS copy (ASYNCcnt-tracked DMA, no VGPR round trip)
// ---------------------------------------------------------------------------
__device__ __forceinline__ void async_copy_b128(unsigned lds_off, const void* gaddr) {
    asm volatile("global_load_async_to_lds_b128 %0, %1, off"
                 :: "v"(lds_off), "v"(gaddr)
                 : "memory");
}
__device__ __forceinline__ void wait_async0() {
    asm volatile("s_wait_asynccnt 0x0" ::: "memory");
}

// ---------------------------------------------------------------------------
// Utility kernels
// ---------------------------------------------------------------------------
__global__ void cvt_f32_bf16(const float* __restrict__ in, __bf16* __restrict__ out, long n) {
    long i = (long)blockIdx.x * blockDim.x + threadIdx.x;
    long stride = (long)gridDim.x * blockDim.x;
    for (; i < n; i += stride) out[i] = (__bf16)in[i];
}

__global__ void copy_f32(const float* __restrict__ in, float* __restrict__ out, long n) {
    long i = (long)blockIdx.x * blockDim.x + threadIdx.x;
    long stride = (long)gridDim.x * blockDim.x;
    for (; i < n; i += stride) out[i] = in[i];
}

__device__ __forceinline__ void blockReduce2(float& a, float& b, float* sm) {
    int t = threadIdx.x;
    sm[t] = a; sm[256 + t] = b;
    __syncthreads();
    for (int off = 128; off > 0; off >>= 1) {
        if (t < off) { sm[t] += sm[t + off]; sm[256 + t] += sm[256 + t + off]; }
        __syncthreads();
    }
    a = sm[0]; b = sm[256];
    __syncthreads();
}

__device__ __forceinline__ float sigmoidf_(float x) { return 1.0f / (1.0f + expf(-x)); }

// ---------------------------------------------------------------------------
// bf16 TN GEMM:  C[m,n] = sum_k A[m,k]*Bm[n,k] + bias[n]
// A: [M,K] row-major bf16; Bm: [N,K] row-major bf16; C: [M,N] f32
// Block tile 128x128, K-stage 64, 256 threads = 8 wave32 waves.
// Double-buffered LDS, async global->LDS fills, 16 WMMAs / wave / stage.
// ---------------------------------------------------------------------------
#define BM 128
#define BN 128
#define BK 64

__global__ __launch_bounds__(256) void gemm_bf16_tn(
    const __bf16* __restrict__ A, const __bf16* __restrict__ Bm,
    const float* __restrict__ bias, float* __restrict__ C,
    int M, int N, int K) {
    __shared__ __bf16 lsA[2][BM * BK];
    __shared__ __bf16 lsB[2][BN * BK];

    const int tid  = threadIdx.x;
    const int lane = tid & 31;
    const int wave = tid >> 5;
    const int waveM = wave >> 1;   // 0..3
    const int waveN = wave & 1;    // 0..1
    const int half = lane >> 4;    // 0/1 half-wave
    const int l16  = lane & 15;

    const int bRow = blockIdx.y * BM;
    const int bCol = blockIdx.x * BN;

    const unsigned aBase = (unsigned)(uintptr_t)(&lsA[0][0]);
    const unsigned bBase = (unsigned)(uintptr_t)(&lsB[0][0]);

    v8f acc[2][4];
#pragma unroll
    for (int mt = 0; mt < 2; mt++)
#pragma unroll
        for (int nt = 0; nt < 4; nt++)
#pragma unroll
            for (int r = 0; r < 8; r++) acc[mt][nt][r] = 0.0f;

    // Issue one full K-stage of async fills (4 x 16B chunks per matrix per thread)
    auto issue_stage = [&](int buf, int kb) {
#pragma unroll
        for (int i = 0; i < 4; i++) {
            int cidx = tid + i * 256;
            int row = cidx >> 3;
            int col = (cidx & 7) << 3;
            unsigned la = aBase + (unsigned)(buf * BM * BK + row * BK + col) * 2u;
            unsigned lb = bBase + (unsigned)(buf * BN * BK + row * BK + col) * 2u;
            async_copy_b128(la, &A[(long)(bRow + row) * K + kb + col]);
            async_copy_b128(lb, &Bm[(long)(bCol + row) * K + kb + col]);
        }
    };

    const int nStages = K / BK;
    issue_stage(0, 0);

    for (int s = 0; s < nStages; s++) {
        const int cur = s & 1;
        wait_async0();        // my async stores for 'cur' landed in LDS
        __syncthreads();      // everyone's landed; prior reads of other buffer done
        if (s + 1 < nStages) issue_stage(1 - cur, (s + 1) * BK);

        const __bf16* sA = &lsA[cur][0];
        const __bf16* sB = &lsB[cur][0];
#pragma unroll
        for (int kk = 0; kk < BK; kk += 32) {
            // Gather all fragments first, then burst the 8 WMMAs.
            v16bf afr[2];
#pragma unroll
            for (int mt = 0; mt < 2; mt++) {
                int m = waveM * 32 + mt * 16 + l16;
                const __bf16* p0 = &sA[m * BK + kk + half * 8];
                union { v8bf h[2]; v16bf v; } u;
                u.h[0] = *(const v8bf*)(p0);
                u.h[1] = *(const v8bf*)(p0 + 16);
                afr[mt] = u.v;
            }
            v16bf bfr[4];
#pragma unroll
            for (int nt = 0; nt < 4; nt++) {
                int n = waveN * 64 + nt * 16 + l16;
                const __bf16* p = &sB[n * BK + kk + half * 8];
                union { v8bf h[2]; v16bf v; } u;
                u.h[0] = *(const v8bf*)(p);
                u.h[1] = *(const v8bf*)(p + 16);
                bfr[nt] = u.v;
            }
#pragma unroll
            for (int nt = 0; nt < 4; nt++)
#pragma unroll
                for (int mt = 0; mt < 2; mt++) {
                    acc[mt][nt] = __builtin_amdgcn_wmma_f32_16x16x32_bf16(
                        false, afr[mt], false, bfr[nt], (short)0, acc[mt][nt],
                        false, false);
                }
        }
        __syncthreads();
    }

    // Epilogue: C/D 16x16 layout -> lane l16 = column, VGPR r = row r (+8 for upper half-wave)
#pragma unroll
    for (int mt = 0; mt < 2; mt++) {
#pragma unroll
        for (int nt = 0; nt < 4; nt++) {
            int n = bCol + waveN * 64 + nt * 16 + l16;
            float bv = bias[n];
            int mBase = bRow + waveM * 32 + mt * 16 + half * 8;
#pragma unroll
            for (int r = 0; r < 8; r++) {
                C[(long)(mBase + r) * N + n] = acc[mt][nt][r] + bv;
            }
        }
    }
}

// ---------------------------------------------------------------------------
// LN over concat(xp, h) row of width 4096, then tanh, write bf16 combined.
// One 256-thread block per batch row.
// ---------------------------------------------------------------------------
__global__ __launch_bounds__(256) void ln1_kernel(
    const float* __restrict__ xp, const float* __restrict__ h,
    const float* __restrict__ g, const float* __restrict__ bb,
    __bf16* __restrict__ comb) {
    __shared__ float sm[512];
    long row = blockIdx.x;
    const float* xr = xp + row * (long)H_SZ;
    const float* hr = h  + row * (long)H_SZ;
    float v[16];
    float s = 0.0f, sq = 0.0f;
#pragma unroll
    for (int i = 0; i < 16; i++) {
        int idx = threadIdx.x + i * 256;            // 0..4095
        float val = (idx < H_SZ) ? xr[idx] : hr[idx - H_SZ];
        v[i] = val; s += val; sq += val * val;
    }
    blockReduce2(s, sq, sm);
    float mean = s * (1.0f / 4096.0f);
    float var  = sq * (1.0f / 4096.0f) - mean * mean;
    float rs   = rsqrtf(var + 1e-5f);
#pragma unroll
    for (int i = 0; i < 16; i++) {
        int idx = threadIdx.x + i * 256;
        float o = tanhf((v[i] - mean) * rs * g[idx] + bb[idx]);
        comb[row * 4096 + idx] = (__bf16)o;
    }
}

// ---------------------------------------------------------------------------
// Fused gate LNs + nonlinearities + cell/hidden LNs.
// gates: [B, 4*H] (f, i, c2, o), one 256-thread block per row.
// out: [2, B, H] (hidden, cell)
// ---------------------------------------------------------------------------
__global__ __launch_bounds__(256) void final_kernel(
    const float* __restrict__ gates, const float* __restrict__ c,
    const float* __restrict__ g_f,  const float* __restrict__ beta_f,
    const float* __restrict__ g_i,  const float* __restrict__ beta_i,
    const float* __restrict__ g_c2, const float* __restrict__ beta_c2,
    const float* __restrict__ g_o,  const float* __restrict__ beta_o,
    const float* __restrict__ g_cn, const float* __restrict__ b_cn,
    const float* __restrict__ g_hn, const float* __restrict__ b_hn,
    float* __restrict__ out) {
    __shared__ float sm[512];
    long row = blockIdx.x;
    const float* gr = gates + row * (long)(4 * H_SZ);
    const float* cr = c + row * (long)H_SZ;

    float pre[4][8];
    float mg[4], rg[4];
#pragma unroll
    for (int gi = 0; gi < 4; gi++) {
        float s = 0.0f, sq = 0.0f;
#pragma unroll
        for (int i = 0; i < 8; i++) {
            float v = gr[gi * H_SZ + threadIdx.x + i * 256];
            pre[gi][i] = v; s += v; sq += v * v;
        }
        blockReduce2(s, sq, sm);
        float m = s * (1.0f / 2048.0f);
        float var = sq * (1.0f / 2048.0f) - m * m;
        mg[gi] = m; rg[gi] = rsqrtf(var + 1e-5f);
    }

    // t = f*c + i*cc
    float tv[8];
    float s = 0.0f, sq = 0.0f;
#pragma unroll
    for (int i = 0; i < 8; i++) {
        int j = threadIdx.x + i * 256;
        float f  = sigmoidf_((pre[0][i] - mg[0]) * rg[0] * g_f[j]  + beta_f[j]);
        float ii = sigmoidf_((pre[1][i] - mg[1]) * rg[1] * g_i[j]  + beta_i[j]);
        float cc = tanhf    ((pre[2][i] - mg[2]) * rg[2] * g_c2[j] + beta_c2[j]);
        float t  = f * cr[j] + ii * cc;
        tv[i] = t; s += t; sq += t * t;
    }
    blockReduce2(s, sq, sm);
    float mc = s * (1.0f / 2048.0f);
    float rc = rsqrtf(sq * (1.0f / 2048.0f) - mc * mc + 1e-5f);

    // next_cell = LN(t); u = o * tanh(next_cell)
    float uv[8];
    s = 0.0f; sq = 0.0f;
#pragma unroll
    for (int i = 0; i < 8; i++) {
        int j = threadIdx.x + i * 256;
        float cell = (tv[i] - mc) * rc * g_cn[j] + b_cn[j];
        out[(long)B_SZ * H_SZ + row * (long)H_SZ + j] = cell;   // next_cell
        float o = sigmoidf_((pre[3][i] - mg[3]) * rg[3] * g_o[j] + beta_o[j]);
        float u = o * tanhf(cell);
        uv[i] = u; s += u; sq += u * u;
    }
    blockReduce2(s, sq, sm);
    float mh = s * (1.0f / 2048.0f);
    float rh = rsqrtf(sq * (1.0f / 2048.0f) - mh * mh + 1e-5f);
#pragma unroll
    for (int i = 0; i < 8; i++) {
        int j = threadIdx.x + i * 256;
        out[row * (long)H_SZ + j] = tanhf((uv[i] - mh) * rh * g_hn[j] + b_hn[j]); // next_hidden
    }
}

// ---------------------------------------------------------------------------
// Launch
// ---------------------------------------------------------------------------
extern "C" void kernel_launch(void* const* d_in, const int* in_sizes, int n_in,
                              void* d_out, int out_size, void* d_ws, size_t ws_size,
                              hipStream_t stream) {
    const float* x      = (const float*)d_in[0];   // [B,1,CIN]
    const float* h      = (const float*)d_in[1];   // [B,H]
    const float* c      = (const float*)d_in[2];   // [B,H]
    const float* W_proj = (const float*)d_in[3];   // [H,CIN]
    const float* b_proj = (const float*)d_in[4];   // [H]
    const float* g_ln   = (const float*)d_in[5];
    const float* b_ln   = (const float*)d_in[6];
    const float* g_cn   = (const float*)d_in[7];
    const float* b_cn   = (const float*)d_in[8];
    const float* g_hn   = (const float*)d_in[9];
    const float* b_hn   = (const float*)d_in[10];
    // gates: f, i, c2, o, each (W, b, g, beta) starting at index 11
    const float* W_g[4]; const float* b_g[4]; const float* gam_g[4]; const float* bet_g[4];
    for (int g = 0; g < 4; g++) {
        W_g[g]   = (const float*)d_in[11 + 4 * g + 0];
        b_g[g]   = (const float*)d_in[11 + 4 * g + 1];
        gam_g[g] = (const float*)d_in[11 + 4 * g + 2];
        bet_g[g] = (const float*)d_in[11 + 4 * g + 3];
    }
    float* out = (float*)d_out;

    const long Bn = B_SZ, Cn = CIN_SZ, Hn = H_SZ;

    // Workspace carve-up (256B aligned regions)
    char* ws = (char*)d_ws;
    size_t off = 0;
    auto carve = [&](size_t bytes) -> void* {
        void* p = ws + off;
        off = (off + bytes + 255) & ~(size_t)255;
        return p;
    };
    __bf16* xb    = (__bf16*)carve(Bn * Cn * 2);           // x in bf16
    __bf16* wpb   = (__bf16*)carve(Hn * Cn * 2);           // W_proj bf16
    __bf16* wgb   = (__bf16*)carve(4 * Hn * 2 * Hn * 2);   // [8192,4096] bf16 gate weights
    float*  biasg = (float*) carve(4 * Hn * 4);            // concat gate biases
    float*  xp    = (float*) carve(Bn * Hn * 4);           // projection output f32
    __bf16* comb  = (__bf16*)carve(Bn * 2 * Hn * 2);       // combined bf16 [B,4096]
    float*  gbuf  = (float*) carve(Bn * 4 * Hn * 4);       // gate pre-activations f32
    (void)ws_size; (void)in_sizes; (void)n_in; (void)out_size;

    // 1) Precision converts
    cvt_f32_bf16<<<4096, 256, 0, stream>>>(x, xb, Bn * Cn);
    cvt_f32_bf16<<<2048, 256, 0, stream>>>(W_proj, wpb, Hn * Cn);
    for (int g = 0; g < 4; g++) {
        cvt_f32_bf16<<<4096, 256, 0, stream>>>(W_g[g], wgb + (long)g * Hn * 2 * Hn, Hn * 2 * Hn);
        copy_f32<<<8, 256, 0, stream>>>(b_g[g], biasg + g * Hn, Hn);
    }

    // 2) xp = x @ W_proj^T + b_proj   (M=8192, N=2048, K=512)
    gemm_bf16_tn<<<dim3(2048 / BN, 8192 / BM), 256, 0, stream>>>(
        xb, wpb, b_proj, xp, 8192, 2048, 512);

    // 3) combined = tanh(LN([xp | h]))  -> bf16
    ln1_kernel<<<B_SZ, 256, 0, stream>>>(xp, h, g_ln, b_ln, comb);

    // 4) all four gates: M=8192, N=8192, K=4096
    gemm_bf16_tn<<<dim3(8192 / BN, 8192 / BM), 256, 0, stream>>>(
        comb, wgb, biasg, gbuf, 8192, 8192, 4096);

    // 5) fused LNs + nonlinearities -> out[2,B,H]
    final_kernel<<<B_SZ, 256, 0, stream>>>(
        gbuf, c,
        gam_g[0], bet_g[0], gam_g[1], bet_g[1],
        gam_g[2], bet_g[2], gam_g[3], bet_g[3],
        g_cn, b_cn, g_hn, b_hn, out);
}
